// ResidualBlock_62208306315888
// MI455X (gfx1250) — matrix operand order repaired and checked by reading, not compiled, not run on hardware
//
#include <hip/hip_runtime.h>
#include <cstdint>
#include <cstddef>

// ---- problem constants ----
#define BB   2
#define LL   1024
#define DD   1024
#define DIN  2048
#define SS   16
#define RR   64
#define KK   4
#define NTOK (BB * LL)          // 2048 token rows
#define NXZ  (2 * DIN)          // 4096
#define NXD  (RR + 2 * SS)      // 96

typedef __attribute__((ext_vector_type(16))) __bf16 bf16x16;
typedef __attribute__((ext_vector_type(8)))  __bf16 bf16x8;
typedef __attribute__((ext_vector_type(8)))  float  floatx8;

__device__ __forceinline__ uint16_t f2bf(float f) {
  union { float f; uint32_t u; } c; c.f = f;
  uint32_t u = c.u;
  uint32_t r = u + 0x7FFFu + ((u >> 16) & 1u);   // round-to-nearest-even
  return (uint16_t)(r >> 16);
}

__device__ __forceinline__ float sigmoidf_(float x) { return 1.0f / (1.0f + __expf(-x)); }

// Build a 16-element bf16 fragment from two 8-element vector loads, fully in SSA
// (no union / alloca => no scratch round-trips).
__device__ __forceinline__ bf16x16 load_frag(const __bf16* p0, const __bf16* p1) {
  bf16x8 lo = *(const bf16x8*)p0;
  bf16x8 hi = *(const bf16x8*)p1;
  return __builtin_shufflevector(lo, hi, 0, 1, 2, 3, 4, 5, 6, 7,
                                         8, 9, 10, 11, 12, 13, 14, 15);
}

// ---------------------------------------------------------------------------
// Weight convert + transpose: out[c*rows + r] = bf16(in[r*cols + c])
// ---------------------------------------------------------------------------
__global__ void convertT_kernel(const float* __restrict__ in, uint16_t* __restrict__ out,
                                int rows, int cols) {
  int idx = blockIdx.x * blockDim.x + threadIdx.x;
  if (idx >= rows * cols) return;
  int r = idx / cols, c = idx - r * cols;
  out[(size_t)c * rows + r] = f2bf(in[idx]);
}

// ---------------------------------------------------------------------------
// RMSNorm over D=1024, one block (256 threads) per token row; outputs bf16.
// ---------------------------------------------------------------------------
__global__ void rmsnorm_kernel(const float* __restrict__ x, const float* __restrict__ w,
                               uint16_t* __restrict__ xn) {
  const int row = blockIdx.x;
  const int tid = threadIdx.x;
  const float4 v = ((const float4*)(x + (size_t)row * DD))[tid];
  float ss = v.x * v.x + v.y * v.y + v.z * v.z + v.w * v.w;
  #pragma unroll
  for (int m = 16; m >= 1; m >>= 1) ss += __shfl_xor(ss, m, 32);
  __shared__ float wsum[8];
  __shared__ float srs;
  if ((tid & 31) == 0) wsum[tid >> 5] = ss;
  __syncthreads();
  if (tid == 0) {
    float t = 0.f;
    #pragma unroll
    for (int i = 0; i < 8; ++i) t += wsum[i];
    srs = rsqrtf(t * (1.0f / DD) + 1e-5f);
  }
  __syncthreads();
  const float rs = srs;
  const float4 wv = ((const float4*)w)[tid];
  uint16_t* o = xn + (size_t)row * DD + tid * 4;
  o[0] = f2bf(v.x * rs * wv.x);
  o[1] = f2bf(v.y * rs * wv.y);
  o[2] = f2bf(v.z * rs * wv.z);
  o[3] = f2bf(v.w * rs * wv.w);
}

// ---------------------------------------------------------------------------
// bf16 WMMA GEMM:  C[M,N] = A[M,K] @ Bt[N,K]^T  (+ optional residual)
// Block = 4 waves; wave w owns 16 rows at (by*4+w)*16 and NT 16-wide N-tiles
// at bx*16*NT. Grid is sized exactly, so the K-loop has ZERO control flow.
// Fragment layouts per CDNA5 ISA 7.12.2 (wave32).
// ---------------------------------------------------------------------------
template <int NT>
__global__ void gemm_bf16_wmma(const __bf16* __restrict__ A, const __bf16* __restrict__ Bt,
                               const float* __restrict__ res, float* __restrict__ C,
                               int M, int N, int K) {
  const int lane  = threadIdx.x & 31;
  const int wave  = threadIdx.x >> 5;
  const int mBase = (blockIdx.y * 4 + wave) * 16;
  const int nBase = blockIdx.x * (16 * NT);

  const int m     = mBase + (lane & 15);
  const int koffA = (lane < 16) ? 0 : 8;    // A: lanes 0-15 hold K {0..7,16..23}
  const int koffB = (lane < 16) ? 0 : 8;    // B: lanes 0-15 hold K {0..7,16..23}
  const int nCol  = nBase + (lane & 15);

  const __bf16* pa = A + (size_t)m * K + koffA;
  const __bf16* pb[NT];
  #pragma unroll
  for (int t = 0; t < NT; ++t) pb[t] = Bt + (size_t)(nCol + t * 16) * K + koffB;

  const floatx8 zero = {0.f, 0.f, 0.f, 0.f, 0.f, 0.f, 0.f, 0.f};
  floatx8 acc[NT];
  #pragma unroll
  for (int t = 0; t < NT; ++t) acc[t] = zero;

  for (int k0 = 0; k0 < K; k0 += 32) {
    const bf16x16 a = load_frag(pa, pa + 16);
    pa += 32;
    #pragma unroll
    for (int t = 0; t < NT; ++t) {
      const bf16x16 b = load_frag(pb[t], pb[t] + 16);
      pb[t] += 32;
      acc[t] = __builtin_amdgcn_wmma_f32_16x16x32_bf16(
          false, a, false, b, (short)0, acc[t], false, false);
    }
  }

  const int rowOff = (lane >> 4) * 8;       // D: VGPR j -> row j + 8*(lane>=16)
  #pragma unroll
  for (int t = 0; t < NT; ++t) {
    #pragma unroll
    for (int j = 0; j < 8; ++j) {
      const size_t idx = (size_t)(mBase + rowOff + j) * N + (nCol + t * 16);
      float v = acc[t][j];
      if (res) v += res[idx];
      C[idx] = v;
    }
  }
}

// ---------------------------------------------------------------------------
// Causal depthwise conv (K=4) + bias + SiLU. xi = first DIN cols of xz.
// ---------------------------------------------------------------------------
__global__ void conv_silu_kernel(const float* __restrict__ xz, const float* __restrict__ cw,
                                 const float* __restrict__ cb, float* __restrict__ xc,
                                 uint16_t* __restrict__ xc_bf) {
  int idx = blockIdx.x * blockDim.x + threadIdx.x;   // over B*L*DIN
  if (idx >= NTOK * DIN) return;
  const int c  = idx & (DIN - 1);
  const int bl = idx >> 11;                 // token row (b*L + l)
  const int l  = bl & (LL - 1);
  float acc = cb[c];
  #pragma unroll
  for (int k = 0; k < KK; ++k) {
    const int ll = l - (KK - 1) + k;
    if (ll >= 0) acc += cw[c * KK + k] * xz[(size_t)(bl - (KK - 1) + k) * NXZ + c];
  }
  const float v = acc * sigmoidf_(acc);
  xc[idx]    = v;
  xc_bf[idx] = f2bf(v);
}

// dt-part of xd (cols 0..63) -> bf16 A matrix for GEMM3
__global__ void dtin_kernel(const float* __restrict__ xd, uint16_t* __restrict__ dtin) {
  int idx = blockIdx.x * blockDim.x + threadIdx.x;   // over NTOK*RR
  if (idx >= NTOK * RR) return;
  const int row = idx >> 6, col = idx & (RR - 1);
  dtin[idx] = f2bf(xd[(size_t)row * NXD + col]);
}

// dt = softplus(dtlin + b_dt), in place
__global__ void softplus_kernel(float* __restrict__ dt, const float* __restrict__ b_dt) {
  int idx = blockIdx.x * blockDim.x + threadIdx.x;   // over NTOK*DIN
  if (idx >= NTOK * DIN) return;
  const float v = dt[idx] + b_dt[idx & (DIN - 1)];
  dt[idx] = (v > 20.f) ? v : log1pf(__expf(v));
}

// ---------------------------------------------------------------------------
// Fused selective scan: lane = (b, c, s); 16 lanes per channel.
//   h = exp(dt*A)*h + dt*xc*B ;  y = sum_s h*C  (shfl reduction)
// Lane s==0 fuses + D_skip*xc, * silu(z), bf16 store of gated y.
// ---------------------------------------------------------------------------
__global__ void scan_kernel(const float* __restrict__ dt, const float* __restrict__ xc,
                            const float* __restrict__ xd, const float* __restrict__ xz,
                            const float* __restrict__ A_log, const float* __restrict__ D_skip,
                            uint16_t* __restrict__ yg) {
  const int tid = blockIdx.x * blockDim.x + threadIdx.x;  // 0 .. B*DIN*S-1
  const int s = tid & (SS - 1);
  const int g = tid >> 4;                  // b*DIN + c
  const int c = g & (DIN - 1);
  const int b = g >> 11;
  const float Ac  = -__expf(A_log[c * SS + s]);
  const float Dsk = D_skip[c];
  float h = 0.f;
  for (int l = 0; l < LL; ++l) {
    const int bl = b * LL + l;
    const float dtv = dt[(size_t)bl * DIN + c];
    const float xcv = xc[(size_t)bl * DIN + c];
    const float Bv  = xd[(size_t)bl * NXD + RR + s];
    const float Cv  = xd[(size_t)bl * NXD + RR + SS + s];
    h = __expf(dtv * Ac) * h + dtv * xcv * Bv;
    float acc = h * Cv;
    acc += __shfl_xor(acc, 1, 32);
    acc += __shfl_xor(acc, 2, 32);
    acc += __shfl_xor(acc, 4, 32);
    acc += __shfl_xor(acc, 8, 32);         // masks <16: stays within 16-lane state group
    if (s == 0) {
      const float zv = xz[(size_t)bl * NXZ + DIN + c];
      const float yv = (acc + Dsk * xcv) * (zv * sigmoidf_(zv));
      yg[(size_t)bl * DIN + c] = f2bf(yv);
    }
  }
}

// ---------------------------------------------------------------------------
extern "C" void kernel_launch(void* const* d_in, const int* in_sizes, int n_in,
                              void* d_out, int out_size, void* d_ws, size_t ws_size,
                              hipStream_t stream) {
  const float* x      = (const float*)d_in[0];
  const float* norm_w = (const float*)d_in[1];
  const float* W_in   = (const float*)d_in[2];
  const float* conv_w = (const float*)d_in[3];
  const float* conv_b = (const float*)d_in[4];
  const float* W_x    = (const float*)d_in[5];
  const float* W_dt   = (const float*)d_in[6];
  const float* b_dt   = (const float*)d_in[7];
  const float* A_log  = (const float*)d_in[8];
  const float* D_skip = (const float*)d_in[9];
  const float* W_out  = (const float*)d_in[10];
  float* out = (float*)d_out;

  size_t off = 0;
  auto wsa = [&](size_t bytes) -> void* {
    off = (off + 255) & ~(size_t)255;
    void* p = (char*)d_ws + off;
    off += bytes;
    return p;
  };
  uint16_t* xn_bf  = (uint16_t*)wsa((size_t)NTOK * DD * 2);
  uint16_t* WinT   = (uint16_t*)wsa((size_t)NXZ * DD * 2);
  uint16_t* WxT    = (uint16_t*)wsa((size_t)NXD * DIN * 2);
  uint16_t* WdtT   = (uint16_t*)wsa((size_t)DIN * RR * 2);
  uint16_t* WoutT  = (uint16_t*)wsa((size_t)DD * DIN * 2);
  float*    xz     = (float*)wsa((size_t)NTOK * NXZ * 4);
  float*    xc     = (float*)wsa((size_t)NTOK * DIN * 4);
  uint16_t* xc_bf  = (uint16_t*)wsa((size_t)NTOK * DIN * 2);
  float*    xd     = (float*)wsa((size_t)NTOK * NXD * 4);
  uint16_t* dtin   = (uint16_t*)wsa((size_t)NTOK * RR * 2);
  float*    dtbuf  = (float*)wsa((size_t)NTOK * DIN * 4);
  uint16_t* yg     = (uint16_t*)wsa((size_t)NTOK * DIN * 2);
  (void)ws_size; (void)in_sizes; (void)n_in; (void)out_size;

  const int T = 256;
  // 1) weights -> bf16, transposed to [N,K]
  convertT_kernel<<<(DD * NXZ + T - 1) / T, T, 0, stream>>>(W_in,  WinT,  DD,  NXZ);
  convertT_kernel<<<(DIN * NXD + T - 1) / T, T, 0, stream>>>(W_x,   WxT,   DIN, NXD);
  convertT_kernel<<<(RR * DIN + T - 1) / T, T, 0, stream>>>(W_dt,  WdtT,  RR,  DIN);
  convertT_kernel<<<(DIN * DD + T - 1) / T, T, 0, stream>>>(W_out, WoutT, DIN, DD);
  // 2) RMSNorm -> bf16 activations
  rmsnorm_kernel<<<NTOK, 256, 0, stream>>>(x, norm_w, xn_bf);
  // 3) GEMM1: xz = xn @ W_in    [2048 x 1024 x 4096], NT=4 -> grid.x=64
  gemm_bf16_wmma<4><<<dim3(NXZ / 64, NTOK / 64), 128, 0, stream>>>(
      (const __bf16*)xn_bf, (const __bf16*)WinT, nullptr, xz, NTOK, NXZ, DD);
  // 4) causal depthwise conv + SiLU
  conv_silu_kernel<<<(NTOK * DIN + T - 1) / T, T, 0, stream>>>(xz, conv_w, conv_b, xc, xc_bf);
  // 5) GEMM2: xd = xc @ W_x     [2048 x 2048 x 96], NT=2 -> grid.x=3 (96/32)
  gemm_bf16_wmma<2><<<dim3(NXD / 32, NTOK / 64), 128, 0, stream>>>(
      (const __bf16*)xc_bf, (const __bf16*)WxT, nullptr, xd, NTOK, NXD, DIN);
  // 6) dt-part -> bf16
  dtin_kernel<<<(NTOK * RR + T - 1) / T, T, 0, stream>>>(xd, dtin);
  // 7) GEMM3: dtlin = dt_part @ W_dt   [2048 x 64 x 2048], NT=4 -> grid.x=32
  gemm_bf16_wmma<4><<<dim3(DIN / 64, NTOK / 64), 128, 0, stream>>>(
      (const __bf16*)dtin, (const __bf16*)WdtT, nullptr, dtbuf, NTOK, DIN, RR);
  // 8) dt = softplus(dtlin + b_dt)
  softplus_kernel<<<(NTOK * DIN + T - 1) / T, T, 0, stream>>>(dtbuf, b_dt);
  // 9) fused selective scan + D-skip + SiLU(z) gate -> bf16 y
  scan_kernel<<<(BB * DIN * SS) / T, T, 0, stream>>>(dtbuf, xc, xd, xz, A_log, D_skip, yg);
  // 10) GEMM4: out = x + yg @ W_out   [2048 x 2048 x 1024], residual fused
  gemm_bf16_wmma<4><<<dim3(DD / 64, NTOK / 64), 128, 0, stream>>>(
      (const __bf16*)yg, (const __bf16*)WoutT, x, out, NTOK, DD, DIN);
}
